// CDMTransformer_0102_78804059947135
// MI455X (gfx1250) — matrix operand order repaired and checked by reading, not compiled, not run on hardware
//
#include <hip/hip_runtime.h>
#include <hip/hip_bf16.h>
#include <math.h>

// ---------------------------------------------------------------------------
// CDM transformer forward for MI455X (gfx1250), wave32 + WMMA bf16 +
// async global->LDS staging (ASYNCcnt, triple buffered) for the GEMM B tiles.
// B=16, S=512, D=1024, H=16, L=4, DH=64.
// ---------------------------------------------------------------------------

typedef __attribute__((ext_vector_type(16))) __bf16 v16bf;
typedef __attribute__((ext_vector_type(8)))  __bf16 v8bf;
typedef __attribute__((ext_vector_type(8)))  float  v8f;
typedef __attribute__((ext_vector_type(4)))  int    v4i;

#define BB   16
#define SS   512
#define DD   1024
#define HH   16
#define LL   4
#define DH   64
#define MTOT (BB * SS)   // 8192
#define EPS  1e-5f

#if defined(__has_builtin)
#if __has_builtin(__builtin_amdgcn_global_load_async_to_lds_b128)
#define ASYNC_LDS 1
#endif
#endif

// ---- helpers --------------------------------------------------------------

__device__ __forceinline__ unsigned short f2bf(float f) {
    unsigned u = __float_as_uint(f);
    u += 0x7fffu + ((u >> 16) & 1u);       // round-to-nearest-even
    return (unsigned short)(u >> 16);
}

__device__ __forceinline__ float wred_max(float v) {
    #pragma unroll
    for (int o = 16; o > 0; o >>= 1) v = fmaxf(v, __shfl_xor(v, o, 32));
    return v;
}
__device__ __forceinline__ float wred_sum(float v) {
    #pragma unroll
    for (int o = 16; o > 0; o >>= 1) v += __shfl_xor(v, o, 32);
    return v;
}

// 16-byte global -> LDS stage, async (CDNA5) when available.
// Builtin signature: (v4i as1*, v4i as3*, imm offset, imm cpol).
__device__ __forceinline__ void stage16(const unsigned short* g, unsigned short* l) {
#ifdef ASYNC_LDS
    __builtin_amdgcn_global_load_async_to_lds_b128(
        (__attribute__((address_space(1))) v4i*)g,
        (__attribute__((address_space(3))) v4i*)l, 0, 0);
#else
    *(v8bf*)l = *(const v8bf*)g;
#endif
}
__device__ __forceinline__ void wait_async_le2() {
#ifdef ASYNC_LDS
#if defined(__has_builtin) && __has_builtin(__builtin_amdgcn_s_wait_asynccnt)
    __builtin_amdgcn_s_wait_asynccnt(2);
#else
    asm volatile("s_wait_asynccnt 0x2" ::: "memory");
#endif
#endif
}
__device__ __forceinline__ void wait_async_le0() {
#ifdef ASYNC_LDS
#if defined(__has_builtin) && __has_builtin(__builtin_amdgcn_s_wait_asynccnt)
    __builtin_amdgcn_s_wait_asynccnt(0);
#else
    asm volatile("s_wait_asynccnt 0x0" ::: "memory");
#endif
#endif
}

// A-fragment (16x32 bf16, M x K): lane (l&15)=M row; lanes<16 hold K 0-15,
// lanes>=16 hold K 16-31.  rowptr = &A[row][kb].
__device__ __forceinline__ v16bf load_a_frag(const unsigned short* rowptr, int lane) {
    union { v16bf v; v8bf h[2]; } u;
    int base = (lane >> 4) * 16;
    u.h[0] = *reinterpret_cast<const v8bf*>(rowptr + base);
    u.h[1] = *reinterpret_cast<const v8bf*>(rowptr + 8 + base);
    return u.v;
}

// B-fragment (32x16 bf16, K x N): lane (l&15)=N col; element i holds
// K = (lane>=16 ? 16 : 0) + i.  p contiguous 16 halfs.
__device__ __forceinline__ v16bf load_b_frag(const unsigned short* p) {
    return *reinterpret_cast<const v16bf*>(p);
}

#define WMMA_BF16(a, b, c) \
    __builtin_amdgcn_wmma_f32_16x16x32_bf16(false, (a), false, (b), (short)0, (c), false, false)

// ---- fp32 -> bf16 weight conversion ---------------------------------------

__global__ void k_cvt_bf16(const float* __restrict__ src,
                           unsigned short* __restrict__ dst, int n) {
    int i = (blockIdx.x * 256 + threadIdx.x) * 4;
    if (i + 3 < n) {
        #pragma unroll
        for (int k = 0; k < 4; ++k) dst[i + k] = f2bf(src[i + k]);
    }
}

// ---- positional encoding + fp32/bf16 activation init ----------------------

__global__ void k_posenc(const float* __restrict__ X,
                         float* __restrict__ Af,
                         unsigned short* __restrict__ Abf) {
    int idx = blockIdx.x * 256 + threadIdx.x;            // < 8388608
    int d = idx & (DD - 1);
    int s = (idx >> 10) & (SS - 1);
    float freq = expf(-(float)(d & ~1) * (9.210340371976184f / (float)DD));
    float ang  = (float)s * freq;
    float pe   = (d & 1) ? cosf(ang) : sinf(ang);
    float v    = X[idx] + pe;
    Af[idx]  = v;
    Abf[idx] = f2bf(v);
}

// ---- GEMM: C[m,n] = sum_k A[m,k] * W[n,k] + bias[n] -----------------------
// Block: 256 thr / 8 waves; wave -> 16(M) x 128(N) tile; WG -> 128 x 128.
// B tile (128 x 32) staged into LDS per k-step via async loads, TRIPLE
// buffered -> single barrier per k-step.  A fragments register-pipelined.
// mode 0: bf16 out [B,H,S,DH]   (Q projection)
// mode 1: bf16 out [B,H,DH,S]   (V projection, transposed per head)
// mode 2: fp32 out row-major    (output projection)

__global__ __launch_bounds__(256) void k_gemm(const unsigned short* __restrict__ A,
                                              const unsigned short* __restrict__ W,
                                              const float* __restrict__ bias,
                                              unsigned short* __restrict__ outBf,
                                              float* __restrict__ outF,
                                              int mode) {
    __shared__ __align__(16) unsigned short bstage[3][128 * 32];   // 3 x 8 KB

    int tid  = threadIdx.x;
    int lane = tid & 31;
    int w    = tid >> 5;
    int n0   = blockIdx.x * 128;
    int m0   = blockIdx.y * 128 + w * 16;
    int nl   = lane & 15;
    int khi  = (lane >> 4) * 16;
    int base = (lane >> 4) * 16;

    // staging: thread -> (column = tid/2, 16-half chunk = (tid&1)*16) of the
    // 128 x 32 B tile; two async b128 per thread per k-step.
    int scol = tid >> 1;
    int sk   = (tid & 1) * 16;
    const unsigned short* gsrc = W + (size_t)(n0 + scol) * DD + sk;
    int soff = scol * 32 + sk;
    unsigned short* sd0 = &bstage[0][soff];
    unsigned short* sd1 = &bstage[1][soff];
    unsigned short* sd2 = &bstage[2][soff];
    const unsigned short* rp0 = &bstage[0][0];
    const unsigned short* rp1 = &bstage[1][0];
    const unsigned short* rp2 = &bstage[2][0];

    v8f acc[8] = {};
    int mrow = m0 + nl;
    const unsigned short* arow = A + (size_t)mrow * DD;

    // prologue: stage k-step 0 into buffer 0, prefetch A fragment 0
    stage16(gsrc, sd0);
    stage16(gsrc + 8, sd0 + 8);
    v8bf a0 = *reinterpret_cast<const v8bf*>(arow + base);
    v8bf a1 = *reinterpret_cast<const v8bf*>(arow + 8 + base);

    const int NS = DD / 32;
    for (int s = 0; s < NS; ++s) {
        int kb = s * 32;
        v8bf ca0 = a0, ca1 = a1;

        if (s + 1 < NS) {
            // stage step s+1 into the "next" buffer, prefetch next A chunk,
            // then wait only for the CURRENT step's 2 async ops (in-order).
            const unsigned short* gs = gsrc + kb + 32;
            stage16(gs, sd1);
            stage16(gs + 8, sd1 + 8);
            a0 = *reinterpret_cast<const v8bf*>(arow + kb + 32 + base);
            a1 = *reinterpret_cast<const v8bf*>(arow + kb + 40 + base);
            wait_async_le2();
        } else {
            wait_async_le0();
        }
        __syncthreads();     // single barrier per step (triple buffering)

        union { v16bf v; v8bf h[2]; } af;
        af.h[0] = ca0; af.h[1] = ca1;
        #pragma unroll
        for (int nt = 0; nt < 8; ++nt) {
            v16bf bf = load_b_frag(rp0 + (nt * 16 + nl) * 32 + khi);
            acc[nt] = WMMA_BF16(af.v, bf, acc[nt]);
        }

        // rotate buffers: cur <- next <- next-next <- cur
        unsigned short* ts = sd0; sd0 = sd1; sd1 = sd2; sd2 = ts;
        const unsigned short* tr = rp0; rp0 = rp1; rp1 = rp2; rp2 = tr;
    }

    #pragma unroll
    for (int nt = 0; nt < 8; ++nt) {
        int ng = n0 + nt * 16 + nl;
        float bv = bias[ng];
        #pragma unroll
        for (int r = 0; r < 8; ++r) {
            int mg = m0 + (lane >> 4) * 8 + r;
            float val = acc[nt][r] + bv;
            if (mode == 2) {
                outF[(size_t)mg * DD + ng] = val;
            } else {
                int b = mg >> 9, s = mg & (SS - 1);
                int h = ng >> 6, dh = ng & (DH - 1);
                if (mode == 0)
                    outBf[(((size_t)(b * HH + h) * SS) + s) * DH + dh] = f2bf(val);
                else
                    outBf[(((size_t)(b * HH + h) * DH) + dh) * SS + s] = f2bf(val);
            }
        }
    }
}

// ---- attention: scores, decay-modulated double softmax, maxout, attn@V ----
// grid.x = S/16 (q tile), grid.y = B*H.  256 threads / 8 waves.

__global__ __launch_bounds__(256) void k_attn(const unsigned short* __restrict__ Q,
                                              const unsigned short* __restrict__ Vt,
                                              unsigned short* __restrict__ AO,
                                              float* __restrict__ scoresOut,
                                              const float* __restrict__ gammaRow) {
    __shared__ float sc[16][SS];                              // 32 KB
    __shared__ __align__(16) unsigned short at[16][SS];       // 16 KB
    __shared__ float red[4][256];                             //  4 KB

    int lane = threadIdx.x & 31;
    int w    = threadIdx.x >> 5;
    int bh   = blockIdx.y;
    int h    = bh & (HH - 1);
    int q0   = blockIdx.x * 16;
    int nl   = lane & 15;
    int khi  = (lane >> 4) * 16;
    const unsigned short* Qh = Q + (size_t)bh * SS * DH;

    // ---- phase A: scores tile 16 x 512 (k == q, kq_same) ----
    {
        int mrow = q0 + nl;
        const unsigned short* arow = Qh + (size_t)mrow * DH;
        #pragma unroll
        for (int t = 0; t < 4; ++t) {
            int j0 = w * 64 + t * 16;
            v8f acc = {};
            #pragma unroll
            for (int ks = 0; ks < 2; ++ks) {
                int kb = ks * 32;
                v16bf af = load_a_frag(arow + kb, lane);
                v16bf bf = load_b_frag(Qh + (size_t)(j0 + nl) * DH + kb + khi);
                acc = WMMA_BF16(af, bf, acc);
            }
            #pragma unroll
            for (int r = 0; r < 8; ++r) {
                int sr = (lane >> 4) * 8 + r;
                sc[sr][j0 + nl] = acc[r] * 0.125f;            // 1/sqrt(64)
            }
        }
    }
    __syncthreads();

    // ---- phase B: per-row softmax chain (wave w owns rows 2w, 2w+1) ----
    float g = -fabsf(gammaRow[h]);
    for (int rr = 0; rr < 2; ++rr) {
        int r  = w * 2 + rr;
        int ig = q0 + r;
        int c0 = lane * 16;
        float sv[16], e[16], cum[16];
        #pragma unroll
        for (int t = 0; t < 16; ++t) sv[t] = sc[r][c0 + t];

        // first masked softmax (mask: j < i strictly)
        float lmax = -1e32f;
        #pragma unroll
        for (int t = 0; t < 16; ++t)
            lmax = fmaxf(lmax, (c0 + t < ig) ? sv[t] : -1e32f);
        float rmax = wred_max(lmax);
        float lsum = 0.f;
        #pragma unroll
        for (int t = 0; t < 16; ++t) {
            float mv = (c0 + t < ig) ? sv[t] : -1e32f;
            e[t] = expf(mv - rmax);
            lsum += e[t];
        }
        float tot = wred_sum(lsum);
        float inv = 1.f / tot;

        // inclusive prefix sum of p (lane-local + wave scan)
        float run = 0.f;
        #pragma unroll
        for (int t = 0; t < 16; ++t) { run += e[t] * inv; cum[t] = run; }
        float incl = run;
        #pragma unroll
        for (int o = 1; o < 32; o <<= 1) {
            float nv = __shfl_up(incl, o, 32);
            if (lane >= o) incl += nv;
        }
        float excl = incl - run;
        float dtot = __shfl(incl, 31, 32);

        // distance-decay modulation + second masked softmax
        float lmax2 = -1e32f;
        #pragma unroll
        for (int t = 0; t < 16; ++t) {
            int j = c0 + t;
            float dc   = excl + cum[t];
            float dist = sqrtf(fmaxf((dtot - dc) * fabsf((float)(ig - j)), 0.f));
            float eff  = expf(dist * g);
            eff = fminf(fmaxf(eff, 1e-5f), 1e5f);
            sv[t] = (j < ig) ? sv[t] * eff : -1e32f;
            lmax2 = fmaxf(lmax2, sv[t]);
        }
        float rmax2 = wred_max(lmax2);
        float lsum2 = 0.f;
        #pragma unroll
        for (int t = 0; t < 16; ++t) { e[t] = expf(sv[t] - rmax2); lsum2 += e[t]; }
        float inv2 = 1.f / wred_sum(lsum2);

        // zero masked, maxout rescale: min(1/max, 5)
        float lamax = 0.f;
        #pragma unroll
        for (int t = 0; t < 16; ++t) {
            float a = (c0 + t < ig) ? e[t] * inv2 : 0.f;
            e[t] = a;
            lamax = fmaxf(lamax, a);
        }
        float amax  = wred_max(lamax);
        float scale = fminf(1.f / amax, 5.f);   // amax==0 -> inf -> 5, attn stays 0
        #pragma unroll
        for (int t = 0; t < 16; ++t) {
            float a = e[t] * scale;
            at[r][c0 + t] = f2bf(a);
            if (scoresOut)
                scoresOut[((size_t)bh * SS + ig) * SS + (c0 + t)] = a;
        }
    }
    __syncthreads();

    // ---- phase C: out = attn @ V (16 x 64), K=512 split across wave pairs ----
    {
        int nt = w & 3, kh = w >> 2;
        v8f acc = {};
        const unsigned short* arow = &at[nl][0];
        int n = nt * 16 + nl;
        const unsigned short* brow = Vt + (size_t)(bh * DH + n) * SS;
        #pragma unroll
        for (int ks = 0; ks < 8; ++ks) {
            int kb = kh * 256 + ks * 32;
            v16bf af = load_a_frag(arow + kb, lane);
            v16bf bf = load_b_frag(brow + kb + khi);
            acc = WMMA_BF16(af, bf, acc);
        }
        if (kh == 1) {
            #pragma unroll
            for (int r = 0; r < 8; ++r) red[nt][r * 32 + lane] = acc[r];
        }
        __syncthreads();
        if (kh == 0) {
            int b = bh >> 4;
            #pragma unroll
            for (int r = 0; r < 8; ++r) {
                float val = acc[r] + red[nt][r * 32 + lane];
                int sg  = q0 + (lane >> 4) * 8 + r;
                int col = h * DH + nt * 16 + nl;
                AO[((size_t)b * SS + sg) * DD + col] = f2bf(val);
            }
        }
    }
}

// ---- residual + LayerNorm (block per row) ---------------------------------
// Yf = LN(X + O) * wt + bs ; optional bf16 copy.  O==nullptr -> LN(X).

__global__ __launch_bounds__(256) void k_ln(const float* __restrict__ X,
                                            const float* __restrict__ O,
                                            const float* __restrict__ wt,
                                            const float* __restrict__ bs,
                                            float* __restrict__ Yf,
                                            unsigned short* __restrict__ Ybf) {
    __shared__ float rb1[8], rb2[8];
    int row  = blockIdx.x;
    int tid  = threadIdx.x;
    int lane = tid & 31, w = tid >> 5;
    const float* xr = X + (size_t)row * DD;
    const float* orow = O ? O + (size_t)row * DD : nullptr;

    float t[4];
    float s = 0.f;
    #pragma unroll
    for (int k = 0; k < 4; ++k) {
        int d = k * 256 + tid;
        float v = xr[d] + (orow ? orow[d] : 0.f);
        t[k] = v; s += v;
    }
    s = wred_sum(s);
    if (lane == 0) rb1[w] = s;
    __syncthreads();
    float tot = 0.f;
    #pragma unroll
    for (int i = 0; i < 8; ++i) tot += rb1[i];
    float mu = tot * (1.f / (float)DD);

    float vs = 0.f;
    #pragma unroll
    for (int k = 0; k < 4; ++k) { float dv = t[k] - mu; vs += dv * dv; }
    vs = wred_sum(vs);
    if (lane == 0) rb2[w] = vs;
    __syncthreads();
    float vtot = 0.f;
    #pragma unroll
    for (int i = 0; i < 8; ++i) vtot += rb2[i];
    float rstd = rsqrtf(vtot * (1.f / (float)DD) + EPS);

    #pragma unroll
    for (int k = 0; k < 4; ++k) {
        int d = k * 256 + tid;
        float y = (t[k] - mu) * rstd * wt[d] + bs[d];
        Yf[(size_t)row * DD + d] = y;
        if (Ybf) Ybf[(size_t)row * DD + d] = f2bf(y);
    }
}

// ---------------------------------------------------------------------------

extern "C" void kernel_launch(void* const* d_in, const int* in_sizes, int n_in,
                              void* d_out, int out_size, void* d_ws, size_t ws_size,
                              hipStream_t stream) {
    (void)in_sizes; (void)n_in; (void)out_size; (void)ws_size;

    const float* x      = (const float*)d_in[0];
    // d_in[1] = lens (unused in eval mode)
    const float* Wq     = (const float*)d_in[2];
    const float* bq     = (const float*)d_in[3];
    const float* Wv     = (const float*)d_in[4];
    const float* bv     = (const float*)d_in[5];
    const float* Wo     = (const float*)d_in[6];
    const float* bo     = (const float*)d_in[7];
    const float* gammas = (const float*)d_in[8];
    const float* ln_w   = (const float*)d_in[9];
    const float* ln_b   = (const float*)d_in[10];
    const float* fln_w  = (const float*)d_in[11];
    const float* fln_b  = (const float*)d_in[12];

    float* out_y      = (float*)d_out;                       // [B,S,D]
    float* out_scores = (float*)d_out + (size_t)MTOT * DD;   // [B,H,S,S]

    // workspace layout
    char* ws = (char*)d_ws;
    const size_t actF  = (size_t)MTOT * DD * sizeof(float);      // 32 MB
    const size_t actB  = (size_t)MTOT * DD * sizeof(short);      // 16 MB
    const size_t wB    = (size_t)LL * DD * DD * sizeof(short);   //  8 MB
    float*          A0    = (float*)(ws);
    float*          A1    = (float*)(ws + actF);
    unsigned short* actbf = (unsigned short*)(ws + 2 * actF);
    unsigned short* Qbf   = (unsigned short*)(ws + 2 * actF + actB);
    unsigned short* Vtbf  = (unsigned short*)(ws + 2 * actF + 2 * actB);
    unsigned short* AObf  = (unsigned short*)(ws + 2 * actF + 3 * actB);
    unsigned short* Wqbf  = (unsigned short*)(ws + 2 * actF + 4 * actB);
    unsigned short* Wvbf  = (unsigned short*)(ws + 2 * actF + 4 * actB + wB);
    unsigned short* Wobf  = (unsigned short*)(ws + 2 * actF + 4 * actB + 2 * wB);

    // weight conversion: 3 tensors of L*D*D = 4M floats each
    {
        int n = LL * DD * DD;
        int grid = n / (256 * 4);
        k_cvt_bf16<<<grid, 256, 0, stream>>>(Wq, Wqbf, n);
        k_cvt_bf16<<<grid, 256, 0, stream>>>(Wv, Wvbf, n);
        k_cvt_bf16<<<grid, 256, 0, stream>>>(Wo, Wobf, n);
    }

    // positional encoding
    k_posenc<<<(MTOT * DD) / 256, 256, 0, stream>>>(x, A0, actbf);

    dim3 gemmGrid(DD / 128, MTOT / 128);     // (8, 64)
    dim3 attnGrid(SS / 16, BB * HH);         // (32, 256)

    float* Acur  = A0;
    float* Anext = A1;
    for (int l = 0; l < LL; ++l) {
        const unsigned short* wq = Wqbf + (size_t)l * DD * DD;
        const unsigned short* wv = Wvbf + (size_t)l * DD * DD;
        const unsigned short* wo = Wobf + (size_t)l * DD * DD;

        // Q projection -> [B,H,S,DH] bf16
        k_gemm<<<gemmGrid, 256, 0, stream>>>(actbf, wq, bq + l * DD, Qbf, nullptr, 0);
        // V projection -> [B,H,DH,S] bf16 (transposed per head)
        k_gemm<<<gemmGrid, 256, 0, stream>>>(actbf, wv, bv + l * DD, Vtbf, nullptr, 1);
        // attention -> AO bf16 [B,S,D]; last layer also writes fp32 attn to d_out
        k_attn<<<attnGrid, 256, 0, stream>>>(Qbf, Vtbf, AObf,
                                             (l == LL - 1) ? out_scores : nullptr,
                                             gammas + l * HH);
        // output projection -> fp32 into Anext
        k_gemm<<<gemmGrid, 256, 0, stream>>>(AObf, wo, bo + l * DD, nullptr, Anext, 2);
        // residual + LN -> Anext (fp32, in place over O) + actbf (bf16)
        k_ln<<<MTOT, 256, 0, stream>>>(Acur, Anext, ln_w + l * DD, ln_b + l * DD,
                                       Anext, actbf);
        float* tmp = Acur; Acur = Anext; Anext = tmp;
    }

    // final LayerNorm -> d_out
    k_ln<<<MTOT, 256, 0, stream>>>(Acur, nullptr, fln_w, fln_b, out_y, nullptr);
}
